// GradientVQ_57080115364777
// MI455X (gfx1250) — matrix operand order repaired and checked by reading, not compiled, not run on hardware
//
#include <hip/hip_runtime.h>
#include <hip/hip_bf16.h>
#include <cfloat>
#include <stdint.h>

#define VQ_D 512
#define VQ_K 8192
#define KT_PER_STAGE 4                         // k-tiles per LDS stage (4 x 16KB)
#define STAGE_HALF (KT_PER_STAGE * 16 * VQ_D)  // f16 elems per stage buffer (64KB)
#define STAGE_BYTES (STAGE_HALF * 2)

typedef __attribute__((ext_vector_type(16))) _Float16 v16h;
typedef __attribute__((ext_vector_type(8)))  _Float16 v8h;
typedef __attribute__((ext_vector_type(4)))  _Float16 v4h;
typedef __attribute__((ext_vector_type(8)))  float    v8f;

// ---- CDNA5 async vmem->LDS copy (ASYNCcnt-tracked), portable inline asm ----
__device__ __forceinline__ void async_copy_b128(uint32_t lds_off, const void* gptr) {
  asm volatile("global_load_async_to_lds_b128 %0, %1, off"
               :: "v"(lds_off), "v"((uint64_t)(uintptr_t)gptr)
               : "memory");
}

__device__ __forceinline__ void wait_async0() {
#if __has_builtin(__builtin_amdgcn_s_wait_asynccnt)
  __builtin_amdgcn_s_wait_asynccnt(0);
#else
  asm volatile("s_wait_asynccnt 0" ::: "memory");
#endif
}

// ---------- kernel 1: codebook fp32 -> f16 (row major) + per-row sum-sq ----------
__global__ void __launch_bounds__(256) vq_prep_cb(const float* __restrict__ cb,
                                                  _Float16* __restrict__ cbh,
                                                  float* __restrict__ csq) {
  const int wave = threadIdx.x >> 5;
  const int lane = threadIdx.x & 31;
  const int k = blockIdx.x * 8 + wave;           // one codebook row per wave
  if (k >= VQ_K) return;
  const float* row = cb + (size_t)k * VQ_D;
  _Float16* orow = cbh + (size_t)k * VQ_D;
  const int base = lane * 16;                    // 32 lanes x 16 = 512
  float s = 0.f;
#pragma unroll
  for (int i = 0; i < 16; i += 4) {
    float4 v = *(const float4*)(row + base + i);
    s += v.x * v.x + v.y * v.y + v.z * v.z + v.w * v.w;
    v4h h = {(_Float16)v.x, (_Float16)v.y, (_Float16)v.z, (_Float16)v.w};
    *(v4h*)(orow + base + i) = h;
  }
#pragma unroll
  for (int m = 16; m >= 1; m >>= 1) s += __shfl_xor(s, m, 32);
  if (lane == 0) csq[k] = s;
}

// ---------- kernel 2: x fp32 -> f16 (row major) ----------
__global__ void __launch_bounds__(256) vq_prep_x(const float* __restrict__ x,
                                                 _Float16* __restrict__ xh,
                                                 int total4) {
  const int t = blockIdx.x * blockDim.x + threadIdx.x;
  if (t >= total4) return;
  float4 v = ((const float4*)x)[t];
  v4h h = {(_Float16)v.x, (_Float16)v.y, (_Float16)v.z, (_Float16)v.w};
  ((v4h*)xh)[t] = h;
}

// ---------- kernel 3: fused f16 WMMA GEMM + running argmin ----------
// 4 waves/block (128 thr), each wave owns a 32-row M-tile (two A-fragment
// sets resident in the extended 1024-VGPR file) so every LDS B-fragment pair
// feeds 4 WMMAs. Codebook k-tiles staged into LDS via double-buffered
// async-to-LDS copies (ASYNCcnt) overlapping the WMMA chains; 64KB stages
// halve the barrier frequency vs 32KB.
__global__ void __launch_bounds__(128, 1) vq_argmin(const _Float16* __restrict__ xh,
                                                    const _Float16* __restrict__ cbh,
                                                    const float* __restrict__ csq,
                                                    int* __restrict__ idx_i,
                                                    float* __restrict__ idx_f,
                                                    int n_rows) {
  __shared__ __align__(32) _Float16 bsm[2][STAGE_HALF];   // 2 x 64KB

  const int tid  = threadIdx.x;
  const int wave = tid >> 5;
  const int lane = tid & 31;
  const int hi = lane >> 4;
  const int lo = lane & 15;

  const int mtiles = n_rows >> 5;                 // 32-row tiles
  int mtile = blockIdx.x * 4 + wave;
  const bool active = (mtile < mtiles);
  if (!active) mtile = mtiles - 1;     // clamp: keep barriers uniform, EXEC full

  const int n0 = mtile * 32;

  // A fragments: lane holds row (n0+lo) / (n0+16+lo); per d-chunk dc the 16
  // f16 are the contiguous run [dc*32 + hi*16, +16).
  const _Float16* xrowA = xh + (size_t)(n0 + lo) * VQ_D;
  const _Float16* xrowB = xh + (size_t)(n0 + 16 + lo) * VQ_D;
  v16h afragA[16], afragB[16];
#pragma unroll
  for (int dc = 0; dc < 16; ++dc) {
    const _Float16* pa = xrowA + dc * 32 + hi * 16;
    v8h a0 = *(const v8h*)(pa);
    v8h a1 = *(const v8h*)(pa + 8);
    afragA[dc] = __builtin_shufflevector(a0, a1, 0, 1, 2, 3, 4, 5, 6, 7,
                                         8, 9, 10, 11, 12, 13, 14, 15);
    const _Float16* pb = xrowB + dc * 32 + hi * 16;
    v8h b0 = *(const v8h*)(pb);
    v8h b1 = *(const v8h*)(pb + 8);
    afragB[dc] = __builtin_shufflevector(b0, b1, 0, 1, 2, 3, 4, 5, 6, 7,
                                         8, 9, 10, 11, 12, 13, 14, 15);
  }

  float bestdA[8], bestdB[8];
  int   bestiA[8], bestiB[8];
#pragma unroll
  for (int r = 0; r < 8; ++r) {
    bestdA[r] = FLT_MAX; bestiA[r] = 0;
    bestdB[r] = FLT_MAX; bestiB[r] = 0;
  }

  const int nstages = (VQ_K / 16) / KT_PER_STAGE;   // 128
  const uint32_t lds0 = (uint32_t)(uintptr_t)(void*)&bsm[0][0];
  const uint32_t lds1 = (uint32_t)(uintptr_t)(void*)&bsm[1][0];

  // prologue: stage 0 into buffer 0 (64KB contiguous span of row-major cbh)
  {
    const char* gsrc = (const char*)cbh;
#pragma unroll
    for (int r = 0; r < 32; ++r) {               // 128 thr x 32 x 16B = 64KB
      const uint32_t off = (uint32_t)(r * 2048 + tid * 16);
      async_copy_b128(lds0 + off, gsrc + off);
    }
  }

  for (int st = 0; st < nstages; ++st) {
    const int buf = st & 1;
    wait_async0();          // my async copies for bsm[buf] are complete
    __syncthreads();        // everyone's copies done AND stage st-1 reads done

    if (st + 1 < nstages) { // overlap next stage copy with this stage's WMMAs
      const char* gsrc = (const char*)cbh + (size_t)(st + 1) * STAGE_BYTES;
      const uint32_t ldsn = (buf ? lds0 : lds1);
#pragma unroll
      for (int r = 0; r < 32; ++r) {
        const uint32_t off = (uint32_t)(r * 2048 + tid * 16);
        async_copy_b128(ldsn + off, gsrc + off);
      }
    }

    const _Float16* sbase = &bsm[buf][0];
#pragma unroll
    for (int sub = 0; sub < KT_PER_STAGE; ++sub) {
      const int kt = st * KT_PER_STAGE + sub;
      // B fragment: lane holds code column (k0+lo); 16 contiguous f16 along d
      // starting at dc*32 + hi*16  (contiguous k order, same as A).
      const _Float16* brow = sbase + sub * (16 * VQ_D) + lo * VQ_D + hi * 16;
      v8f accA0 = {}, accA1 = {}, accB0 = {}, accB1 = {};
#pragma unroll
      for (int dc = 0; dc < 16; dc += 2) {     // each B pair feeds 4 WMMAs
        v16h b0 = *(const v16h*)(brow + dc * 32);
        v16h b1 = *(const v16h*)(brow + dc * 32 + 32);
        accA0 = __builtin_amdgcn_wmma_f32_16x16x32_f16(false, afragA[dc], false, b0,
                                                       (short)0, accA0, false, false);
        accB0 = __builtin_amdgcn_wmma_f32_16x16x32_f16(false, afragB[dc], false, b0,
                                                       (short)0, accB0, false, false);
        accA1 = __builtin_amdgcn_wmma_f32_16x16x32_f16(false, afragA[dc + 1], false, b1,
                                                       (short)0, accA1, false, false);
        accB1 = __builtin_amdgcn_wmma_f32_16x16x32_f16(false, afragB[dc + 1], false, b1,
                                                       (short)0, accB1, false, false);
      }
      const int kcode = kt * 16 + lo;
      const float cs = csq[kcode];
#pragma unroll
      for (int r = 0; r < 8; ++r) {            // C VGPR r, lane: row = r + 8*hi
        float dA = cs - 2.0f * (accA0[r] + accA1[r]);
        if (dA < bestdA[r]) { bestdA[r] = dA; bestiA[r] = kcode; }
        float dB = cs - 2.0f * (accB0[r] + accB1[r]);
        if (dB < bestdB[r]) { bestdB[r] = dB; bestiB[r] = kcode; }
      }
    }
  }

  // argmin across the 16 code-columns within each half-wave; lower index wins ties
#pragma unroll
  for (int r = 0; r < 8; ++r) {
#pragma unroll
    for (int s = 1; s < 16; s <<= 1) {
      float od = __shfl_xor(bestdA[r], s, 16);
      int   oi = __shfl_xor(bestiA[r], s, 16);
      if (od < bestdA[r] || (od == bestdA[r] && oi < bestiA[r])) {
        bestdA[r] = od; bestiA[r] = oi;
      }
      od = __shfl_xor(bestdB[r], s, 16);
      oi = __shfl_xor(bestiB[r], s, 16);
      if (od < bestdB[r] || (od == bestdB[r] && oi < bestiB[r])) {
        bestdB[r] = od; bestiB[r] = oi;
      }
    }
  }
  if (active && lo == 0) {
#pragma unroll
    for (int r = 0; r < 8; ++r) {
      int nA = n0 + hi * 8 + r;
      idx_i[nA] = bestiA[r];
      idx_f[nA] = (float)bestiA[r];
      int nB = n0 + 16 + hi * 8 + r;
      idx_i[nB] = bestiB[r];
      idx_f[nB] = (float)bestiB[r];
    }
  }
}

// ---------- kernel 4: gather quantized rows (fp32) + per-block loss partials ----------
__global__ void __launch_bounds__(256) vq_gather_loss(const float* __restrict__ x,
                                                      const float* __restrict__ cb,
                                                      const int* __restrict__ idx,
                                                      float* __restrict__ out_q,
                                                      float* __restrict__ partial,
                                                      int total) {
  float s = 0.f;
  for (int t = blockIdx.x * blockDim.x + threadIdx.x; t < total;
       t += gridDim.x * blockDim.x) {
    const int n = t >> 9;            // D = 512
    const int d = t & (VQ_D - 1);
    float q = cb[(size_t)idx[n] * VQ_D + d];
    float diff = q - x[t];
    s += diff * diff;
    out_q[t] = q;                    // straight-through value == quantized
  }
  __shared__ float red[8];
#pragma unroll
  for (int m = 16; m >= 1; m >>= 1) s += __shfl_xor(s, m, 32);
  const int lane = threadIdx.x & 31, wave = threadIdx.x >> 5;
  if (lane == 0) red[wave] = s;
  __syncthreads();
  if (threadIdx.x == 0) {
    float t2 = 0.f;
#pragma unroll
    for (int w = 0; w < 8; ++w) t2 += red[w];
    partial[blockIdx.x] = t2;        // deterministic (no atomics)
  }
}

// ---------- kernel 5: final deterministic loss reduce ----------
__global__ void __launch_bounds__(256) vq_final(const float* __restrict__ partial,
                                                int nparts,
                                                float* __restrict__ out_loss,
                                                float inv_total) {
  float s = 0.f;
  for (int i = threadIdx.x; i < nparts; i += blockDim.x) s += partial[i];
  __shared__ float red[8];
#pragma unroll
  for (int m = 16; m >= 1; m >>= 1) s += __shfl_xor(s, m, 32);
  const int lane = threadIdx.x & 31, wave = threadIdx.x >> 5;
  if (lane == 0) red[wave] = s;
  __syncthreads();
  if (threadIdx.x == 0) {
    float t = 0.f;
#pragma unroll
    for (int w = 0; w < 8; ++w) t += red[w];
    // vq_loss = mse(q, x) + 0.25 * mse(x, q) = 1.25 * mse
    *out_loss = 1.25f * t * inv_total;
  }
}

extern "C" void kernel_launch(void* const* d_in, const int* in_sizes, int n_in,
                              void* d_out, int out_size, void* d_ws, size_t ws_size,
                              hipStream_t stream) {
  const float* x  = (const float*)d_in[0];   // [8,4096,512] f32
  const float* cb = (const float*)d_in[1];   // [8192,512]   f32
  const int n_rows = in_sizes[0] / VQ_D;     // 32768
  const int total  = n_rows * VQ_D;

  // workspace layout (all sub-buffers naturally 256B+ aligned)
  char* ws = (char*)d_ws;
  _Float16* xh  = (_Float16*)ws;  ws += (size_t)total * sizeof(_Float16);        // 32 MB
  _Float16* cbh = (_Float16*)ws;  ws += (size_t)VQ_K * VQ_D * sizeof(_Float16);  //  8 MB
  float* csq    = (float*)ws;     ws += (size_t)VQ_K * sizeof(float);
  int*   idx_i  = (int*)ws;       ws += (size_t)n_rows * sizeof(int);
  float* partial = (float*)ws;

  // output: [quantized (N*D) | vq_loss (1) | indices (N)]
  float* out_q    = (float*)d_out;
  float* out_loss = out_q + total;
  float* out_idx  = out_loss + 1;

  vq_prep_cb<<<VQ_K / 8, 256, 0, stream>>>(cb, cbh, csq);
  vq_prep_x<<<(total / 4 + 255) / 256, 256, 0, stream>>>(x, xh, total / 4);
  vq_argmin<<<(n_rows / 32 + 3) / 4, 128, 0, stream>>>(xh, cbh, csq, idx_i,
                                                       out_idx, n_rows);
  const int GBLK = 4096;
  vq_gather_loss<<<GBLK, 256, 0, stream>>>(x, cb, idx_i, out_q, partial, total);
  vq_final<<<1, 256, 0, stream>>>(partial, GBLK, out_loss, 1.0f / (float)total);
}
